// SelfAttention_80152679678642
// MI455X (gfx1250) — matrix operand order repaired and checked by reading, not compiled, not run on hardware
//
#include <hip/hip_runtime.h>

typedef __bf16 bf16;
typedef __attribute__((ext_vector_type(16))) __bf16 v16bf;
typedef __attribute__((ext_vector_type(8)))  __bf16 v8bf;
typedef __attribute__((ext_vector_type(8)))  float  v8f;
typedef __attribute__((ext_vector_type(4)))  unsigned int u32x4;
typedef __attribute__((ext_vector_type(4)))  int i32x4;
typedef __attribute__((ext_vector_type(8)))  int i32x8;

static __device__ __forceinline__ v8f zero_v8f() {
  v8f z;
#pragma unroll
  for (int i = 0; i < 8; ++i) z[i] = 0.0f;
  return z;
}

// max across the 16-lane key group using DPP only (no LDS traffic)
static __device__ __forceinline__ float max16_dpp(float x) {
  int t;
  t = __builtin_amdgcn_update_dpp(0, __float_as_int(x), 0xB1, 0xF, 0xF, true);  // quad xor1
  x = fmaxf(x, __int_as_float(t));
  t = __builtin_amdgcn_update_dpp(0, __float_as_int(x), 0x4E, 0xF, 0xF, true);  // quad xor2
  x = fmaxf(x, __int_as_float(t));
  t = __builtin_amdgcn_update_dpp(0, __float_as_int(x), 0x141, 0xF, 0xF, true); // half mirror
  x = fmaxf(x, __int_as_float(t));
  t = __builtin_amdgcn_update_dpp(0, __float_as_int(x), 0x140, 0xF, 0xF, true); // row mirror
  x = fmaxf(x, __int_as_float(t));
  return x;
}

// ---------------------------------------------------------------------------
// TDM: async 2-D bf16 tile load Global -> LDS with row padding.
// ---------------------------------------------------------------------------
static __device__ __forceinline__ void tdm_load_2d(const void* lds_dst,
                                                   const void* gsrc,
                                                   unsigned tile_d0, unsigned tile_d1,
                                                   unsigned long long row_stride,
                                                   int pad_interval, int pad_amount) {
  unsigned long long ga = (unsigned long long)gsrc;
  unsigned lds_addr = (unsigned)(unsigned long long)lds_dst;  // low 32 bits = LDS offset
  u32x4 g0;
  g0[0] = 1u;                                                  // count=1, user mode
  g0[1] = lds_addr;
  g0[2] = (unsigned)ga;                                        // global_addr[31:0]
  g0[3] = ((unsigned)(ga >> 32) & 0x01FFFFFFu) | 0x80000000u;  // addr[56:32] | type=2
  i32x8 g1;
  g1[0] = (1 << 16) | (1 << 20) | (pad_interval << 22) | (pad_amount << 25);
  unsigned td0 = tile_d0, td1 = tile_d1;
  g1[1] = (int)((td0 & 0xFFFFu) << 16);
  g1[2] = (int)((td0 >> 16) | ((td1 & 0xFFFFu) << 16));
  g1[3] = (int)((td1 >> 16) | (tile_d0 << 16));
  g1[4] = (int)tile_d1;
  g1[5] = (int)(unsigned)row_stride;
  g1[6] = (int)((row_stride >> 32) & 0xFFFFull);
  g1[7] = 0;
  i32x4 z4;
  z4[0] = 0; z4[1] = 0; z4[2] = 0; z4[3] = 0;
  i32x8 z8;
#pragma unroll
  for (int i = 0; i < 8; ++i) z8[i] = 0;
  __builtin_amdgcn_tensor_load_to_lds(g0, g1, z4, z4, z8, 0);
}

// ---------------------------------------------------------------------------
// fp32 -> bf16 (scaled) elementwise
// ---------------------------------------------------------------------------
__global__ __launch_bounds__(256) void cvt_f32_bf16(const float* __restrict__ src,
                                                    bf16* __restrict__ dst,
                                                    int n, float scale) {
  int i = blockIdx.x * 256 + threadIdx.x;
  if (i < n) dst[i] = (bf16)(src[i] * scale);
}

// ---------------------------------------------------------------------------
// fmap [b][512][4096] f32 -> fmapT [b][4096][512] bf16 (tiled transpose)
// ---------------------------------------------------------------------------
__global__ __launch_bounds__(256) void transpose_cvt(const float* __restrict__ src,
                                                     bf16* __restrict__ dst) {
  __shared__ float tile[32][33];
  const int tx = threadIdx.x, ty = threadIdx.y;  // 32 x 8
  const int x0 = blockIdx.x * 32;
  const int c0 = blockIdx.y * 32;
  const int b  = blockIdx.z;
#pragma unroll
  for (int i = 0; i < 4; ++i)
    tile[ty + i * 8][tx] = src[((long)(b * 512 + c0 + ty + i * 8) * 4096) + x0 + tx];
  __syncthreads();
#pragma unroll
  for (int i = 0; i < 4; ++i)
    dst[((long)(b * 4096 + x0 + ty + i * 8) * 512) + c0 + tx] =
        (bf16)tile[tx][ty + i * 8];
}

// ---------------------------------------------------------------------------
// im2col for stride-2 2x2 conv, output [b][m=1024][k=2048] (k fastest)
// ---------------------------------------------------------------------------
__global__ __launch_bounds__(256) void im2col_s2k2(const float* __restrict__ fmap,
                                                   bf16* __restrict__ im) {
  int idx = blockIdx.x * 256 + threadIdx.x;  // total 4*1024*2048
  int kk = idx & 2047;
  int m  = (idx >> 11) & 1023;
  int b  = idx >> 21;
  int y2 = m & 31, x2 = m >> 5;
  int j = kk & 1, i = (kk >> 1) & 1, c = kk >> 2;
  float v = fmap[(((b * 512 + c) * 64) + (2 * x2 + i)) * 64 + (2 * y2 + j)];
  im[idx] = (bf16)v;
}

// ---------------------------------------------------------------------------
// Tiled bf16 GEMM: C = A * B,  A [M][K], B [N][K] row-major.
// TDM double-buffered staging; K-loop unrolled x2 with buffer parity static.
// 256 threads = 8 waves (2x4); tile 128x256x32; each wave 64x64 = 16 WMMA/step.
// ---------------------------------------------------------------------------
template <bool OUT_BF16, bool TRANS_C>
__global__ __launch_bounds__(256) void gemm_bf16(const bf16* __restrict__ A,
                                                 const bf16* __restrict__ B,
                                                 void* __restrict__ Cv,
                                                 int M, int N, int K,
                                                 long strideA, long strideB, long strideC) {
  constexpr int BM = 128, BN = 256, BK = 32, LDSK = BK + 8;  // rows 80B, 16B aligned
  __shared__ bf16 sA[2][BM * LDSK];
  __shared__ bf16 sB[2][BN * LDSK];

  const int tid  = threadIdx.x;
  const int lane = tid & 31;
  const int wid  = tid >> 5;
  const int m0 = blockIdx.x * BM;
  const int n0 = blockIdx.y * BN;
  const int bz = blockIdx.z;

  A += (long)bz * strideA;
  B += (long)bz * strideB;

  const int wm = (wid & 1) * 64;
  const int wn = (wid >> 1) * 64;
  const int row16 = lane & 15;
  const int hi    = (lane >= 16) ? 1 : 0;
  const int aklo  = hi ? 8 : 0;
  const int bkb   = hi ? 8 : 0;

  v8f acc[4][4];
#pragma unroll
  for (int mi = 0; mi < 4; ++mi)
#pragma unroll
    for (int ni = 0; ni < 4; ++ni) acc[mi][ni] = zero_v8f();

  // pad codes: rows 32 elems = 64B = 16 DW -> interval 3; pad 8 elems = 4 DW -> 3
  auto issue = [&](int k0, int buf) {
    tdm_load_2d(&sA[buf][0], A + (long)m0 * K + k0, BK, BM, (unsigned long long)K, 3, 3);
    tdm_load_2d(&sB[buf][0], B + (long)n0 * K + k0, BK, BN, (unsigned long long)K, 3, 3);
  };
  auto compute = [&](int cur) {
    v16bf afrag[4], bfrag[4];
#pragma unroll
    for (int mi = 0; mi < 4; ++mi) {
      const bf16* p = &sA[cur][(wm + mi * 16 + row16) * LDSK];
      *(v8bf*)&afrag[mi]       = *(const v8bf*)(p + aklo);
      *((v8bf*)&afrag[mi] + 1) = *(const v8bf*)(p + aklo + 16);
    }
#pragma unroll
    for (int ni = 0; ni < 4; ++ni) {
      const bf16* p = &sB[cur][(wn + ni * 16 + row16) * LDSK];
      *(v8bf*)&bfrag[ni]       = *(const v8bf*)(p + bkb);
      *((v8bf*)&bfrag[ni] + 1) = *(const v8bf*)(p + bkb + 16);
    }
    __builtin_amdgcn_sched_barrier(0);  // batch all ds_loads before the WMMA block
#pragma unroll
    for (int mi = 0; mi < 4; ++mi)
#pragma unroll
      for (int ni = 0; ni < 4; ++ni)
        acc[mi][ni] = __builtin_amdgcn_wmma_f32_16x16x32_bf16(
            false, afrag[mi], false, bfrag[ni], (short)0, acc[mi][ni], false, false);
  };

  if (wid == 0) issue(0, 0);
  const int nsteps = K / BK;  // even for all our shapes
  for (int step = 0; step < nsteps; step += 2) {
    __syncthreads();
    if (wid == 0) issue((step + 1) * BK, 1);
    __builtin_amdgcn_s_wait_tensorcnt(2);  // buf0 pair landed
    __syncthreads();
    compute(0);
    __syncthreads();
    if (step + 2 < nsteps) {
      if (wid == 0) issue((step + 2) * BK, 0);
      __builtin_amdgcn_s_wait_tensorcnt(2);  // buf1 pair landed
    } else {
      __builtin_amdgcn_s_wait_tensorcnt(0);
    }
    __syncthreads();
    compute(1);
  }

  const int mrow = hi * 8;
  if (TRANS_C) {
    bf16* C = (bf16*)Cv + (long)bz * strideC;
#pragma unroll
    for (int mi = 0; mi < 4; ++mi)
#pragma unroll
      for (int ni = 0; ni < 4; ++ni) {
        v8bf pk;
#pragma unroll
        for (int v = 0; v < 8; ++v) pk[v] = (bf16)acc[mi][ni][v];
        long n = n0 + wn + ni * 16 + row16;
        long m = m0 + wm + mi * 16 + mrow;
        *(v8bf*)&C[n * M + m] = pk;
      }
  } else if (OUT_BF16) {
    bf16* C = (bf16*)Cv + (long)bz * strideC;
#pragma unroll
    for (int mi = 0; mi < 4; ++mi)
#pragma unroll
      for (int ni = 0; ni < 4; ++ni)
#pragma unroll
        for (int v = 0; v < 8; ++v) {
          long m = m0 + wm + mi * 16 + v + mrow;
          long n = n0 + wn + ni * 16 + row16;
          C[m * N + n] = (bf16)acc[mi][ni][v];
        }
  } else {
    float* C = (float*)Cv + (long)bz * strideC;
#pragma unroll
    for (int mi = 0; mi < 4; ++mi)
#pragma unroll
      for (int ni = 0; ni < 4; ++ni)
#pragma unroll
        for (int v = 0; v < 8; ++v) {
          long m = m0 + wm + mi * 16 + v + mrow;
          long n = n0 + wn + ni * 16 + row16;
          C[m * N + n] = acc[mi][ni][v];
        }
  }
}

// ---------------------------------------------------------------------------
// Flash attention (exp2 domain; 0.125*log2e folded into Wq => into Qt).
// Qt [b][4096][512], Kt [b][1024][512], Vt [b][512][1024] (pre-transposed).
// TDM double-buffered K/V tiles; row max via DPP; row sum via ones-WMMA.
// Grid (64,32), 128 threads = 4 waves, 16 queries/wave, 32 key tiles of 32.
// ---------------------------------------------------------------------------
__global__ __launch_bounds__(128) void flash_attn(const bf16* __restrict__ Qt,
                                                  const bf16* __restrict__ Kt,
                                                  const bf16* __restrict__ Vt,
                                                  const float* __restrict__ null_kv,
                                                  bf16* __restrict__ Og) {
  constexpr int LK = 72;  // 64 + 8 elems, rows 144B
  constexpr int LV = 40;  // 32 + 8 elems, rows 80B
  constexpr int LP = 40;
  __shared__ bf16 sK[2][32 * LK];   // [key][dh]
  __shared__ bf16 sV[2][64 * LV];   // [dh][key]
  __shared__ bf16 sP[4 * 16 * LP];  // per-wave P [row][key]

  const int tid  = threadIdx.x;
  const int lane = tid & 31;
  const int wid  = tid >> 5;
  const int qb = blockIdx.x;
  const int bh = blockIdx.y;
  const int b  = bh >> 3;
  const int h  = bh & 7;
  const int qbase = qb * 64 + wid * 16;
  const int row16 = lane & 15;
  const int hi    = (lane >= 16) ? 1 : 0;
  const int aklo  = hi ? 8 : 0;
  const int bkb   = hi ? 8 : 0;

  // Q fragments, held in registers for the whole kernel
  const bf16* qp = Qt + ((long)(b * 4096 + qbase + row16) * 512) + h * 64;
  v16bf qf[2];
  *(v8bf*)&qf[0]       = *(const v8bf*)(qp + aklo);
  *((v8bf*)&qf[0] + 1) = *(const v8bf*)(qp + aklo + 16);
  *(v8bf*)&qf[1]       = *(const v8bf*)(qp + 32 + aklo);
  *((v8bf*)&qf[1] + 1) = *(const v8bf*)(qp + 32 + aklo + 16);

  v16bf ones;
#pragma unroll
  for (int i = 0; i < 16; ++i) ones[i] = (bf16)1.0f;

  // null token: lane L computes s_null (log2 domain) for row L&15
  const float* nk = null_kv + h * 64;
  const float* nv = null_kv + 512 + h * 64;
  float snull = 0.0f;
  {
    const bf16* qrow = Qt + ((long)(b * 4096 + qbase + row16) * 512) + h * 64;
#pragma unroll 8
    for (int d = 0; d < 64; ++d) snull += (float)qrow[d] * nk[d];
  }
  float m[8], l[8];
  v8f o[4];
#pragma unroll
  for (int v = 0; v < 8; ++v) {
    m[v] = __shfl(snull, v + hi * 8, 32);
    l[v] = 1.0f;
  }
#pragma unroll
  for (int c = 0; c < 4; ++c) {
    float nvv = nv[c * 16 + row16];
#pragma unroll
    for (int v = 0; v < 8; ++v) o[c][v] = nvv;
  }

  // TDM K/V tile issue: K rows 64 elems=128B=32DW -> interval 4; V rows 64B -> 3
  auto issue = [&](int kt, int buf) {
    tdm_load_2d(&sK[buf][0], Kt + ((long)(b * 1024 + kt) * 512) + h * 64,
                64, 32, 512ull, 4, 3);
    tdm_load_2d(&sV[buf][0], Vt + ((long)(b * 512 + h * 64) * 1024) + kt,
                32, 64, 1024ull, 3, 3);
  };

  auto tile_step = [&](int cur) {
    // all K fragments, then 4 QK WMMAs back-to-back
    v16bf kb00, kb01, kb10, kb11;
    {
      const bf16* kp0 = &sK[cur][(row16) * LK];
      const bf16* kp1 = &sK[cur][(16 + row16) * LK];
      *(v8bf*)&kb00       = *(const v8bf*)(kp0 + bkb);
      *((v8bf*)&kb00 + 1) = *(const v8bf*)(kp0 + bkb + 16);
      *(v8bf*)&kb01       = *(const v8bf*)(kp0 + 32 + bkb);
      *((v8bf*)&kb01 + 1) = *(const v8bf*)(kp0 + 32 + bkb + 16);
      *(v8bf*)&kb10       = *(const v8bf*)(kp1 + bkb);
      *((v8bf*)&kb10 + 1) = *(const v8bf*)(kp1 + bkb + 16);
      *(v8bf*)&kb11       = *(const v8bf*)(kp1 + 32 + bkb);
      *((v8bf*)&kb11 + 1) = *(const v8bf*)(kp1 + 32 + bkb + 16);
    }
    __builtin_amdgcn_sched_barrier(0);  // batch the 8 ds_loads before WMMAs
    v8f s0 = zero_v8f(), s1 = zero_v8f();
    s0 = __builtin_amdgcn_wmma_f32_16x16x32_bf16(false, qf[0], false, kb00, (short)0, s0, false, false);
    s1 = __builtin_amdgcn_wmma_f32_16x16x32_bf16(false, qf[0], false, kb10, (short)0, s1, false, false);
    s0 = __builtin_amdgcn_wmma_f32_16x16x32_bf16(false, qf[1], false, kb01, (short)0, s0, false, false);
    s1 = __builtin_amdgcn_wmma_f32_16x16x32_bf16(false, qf[1], false, kb11, (short)0, s1, false, false);

    // online softmax: max via DPP, p = exp2(s - m)
    float f[8];
#pragma unroll
    for (int v = 0; v < 8; ++v) {
      float mt = max16_dpp(fmaxf(s0[v], s1[v]));
      float mn = fmaxf(m[v], mt);
      f[v] = exp2f(m[v] - mn);
      m[v] = mn;
      s0[v] = exp2f(s0[v] - mn);
      s1[v] = exp2f(s1[v] - mn);
    }
#pragma unroll
    for (int c = 0; c < 4; ++c)
#pragma unroll
      for (int v = 0; v < 8; ++v) o[c][v] *= f[v];

    // transpose P through per-wave LDS (C layout -> A layout)
    bf16* pp = &sP[wid * 16 * LP];
#pragma unroll
    for (int v = 0; v < 8; ++v) {
      pp[(v + hi * 8) * LP + row16]      = (bf16)s0[v];
      pp[(v + hi * 8) * LP + 16 + row16] = (bf16)s1[v];
    }
    asm volatile("s_wait_dscnt 0" ::: "memory");
    v16bf pf;
    const bf16* pr = pp + row16 * LP;
    *(v8bf*)&pf       = *(const v8bf*)(pr + aklo);
    *((v8bf*)&pf + 1) = *(const v8bf*)(pr + aklo + 16);

    // all V fragments, then ones-WMMA (row sums) + 4 PV WMMAs
    v16bf vb[4];
#pragma unroll
    for (int c = 0; c < 4; ++c) {
      const bf16* vp = &sV[cur][(c * 16 + row16) * LV];
      *(v8bf*)&vb[c]       = *(const v8bf*)(vp + bkb);
      *((v8bf*)&vb[c] + 1) = *(const v8bf*)(vp + bkb + 16);
    }
    __builtin_amdgcn_sched_barrier(0);
    v8f psum = zero_v8f();
    psum = __builtin_amdgcn_wmma_f32_16x16x32_bf16(false, pf, false, ones, (short)0, psum, false, false);
#pragma unroll
    for (int c = 0; c < 4; ++c)
      o[c] = __builtin_amdgcn_wmma_f32_16x16x32_bf16(false, pf, false, vb[c], (short)0, o[c], false, false);
#pragma unroll
    for (int v = 0; v < 8; ++v) l[v] = l[v] * f[v] + psum[v];
  };

  if (wid == 0) issue(0, 0);
  for (int t = 0; t < 32; t += 2) {
    __syncthreads();
    if (wid == 0) issue((t + 1) * 32, 1);
    __builtin_amdgcn_s_wait_tensorcnt(2);
    __syncthreads();
    tile_step(0);
    __syncthreads();
    if (t + 2 < 32) {
      if (wid == 0) issue((t + 2) * 32, 0);
      __builtin_amdgcn_s_wait_tensorcnt(2);
    } else {
      __builtin_amdgcn_s_wait_tensorcnt(0);
    }
    __syncthreads();
    tile_step(1);
  }

  // epilogue: O /= l, store bf16 [b][query][h*64+dh]
#pragma unroll
  for (int c = 0; c < 4; ++c)
#pragma unroll
    for (int v = 0; v < 8; ++v) {
      float val = o[c][v] / l[v];
      int q = qbase + v + hi * 8;
      Og[((long)(b * 4096 + q) * 512) + h * 64 + c * 16 + row16] = (bf16)val;
    }
}

// ---------------------------------------------------------------------------
// launcher
// ---------------------------------------------------------------------------
extern "C" void kernel_launch(void* const* d_in, const int* in_sizes, int n_in,
                              void* d_out, int out_size, void* d_ws, size_t ws_size,
                              hipStream_t stream) {
  const float* fmap   = (const float*)d_in[0];
  const float* Wq     = (const float*)d_in[1];
  const float* Wk     = (const float*)d_in[2];
  const float* Wv     = (const float*)d_in[3];
  const float* nullkv = (const float*)d_in[4];
  const float* Wout   = (const float*)d_in[5];
  float* out = (float*)d_out;

  char* ws = (char*)d_ws;
  size_t off = 0;
  auto alloc = [&](size_t bytes) -> void* {
    void* p = ws + off;
    off += (bytes + 255) & ~(size_t)255;
    return p;
  };
  bf16* fmapT = (bf16*)alloc(4ull * 4096 * 512 * 2);   // [b][xy][c]
  bf16* WqB   = (bf16*)alloc(512ull * 512 * 2);
  bf16* WkB   = (bf16*)alloc(512ull * 2048 * 2);
  bf16* WvB   = (bf16*)alloc(512ull * 2048 * 2);
  bf16* WoutB = (bf16*)alloc(512ull * 512 * 2);
  bf16* imT   = (bf16*)alloc(4ull * 1024 * 2048 * 2);  // [b][m][k]
  bf16* Qt    = (bf16*)alloc(4ull * 4096 * 512 * 2);   // [b][xy][o]
  bf16* Ktb   = (bf16*)alloc(4ull * 1024 * 512 * 2);   // [b][key][o]
  bf16* Vtb   = (bf16*)alloc(4ull * 512 * 1024 * 2);   // [b][o][key]
  bf16* AO    = (bf16*)alloc(4ull * 4096 * 512 * 2);   // [b][xy][o]

  const float qscale = 0.125f * 1.44269504088896340736f;  // scale * log2(e)
  auto cvt = [&](const float* s, bf16* d, int n, float sc) {
    cvt_f32_bf16<<<(n + 255) / 256, 256, 0, stream>>>(s, d, n, sc);
  };
  {
    dim3 g(4096 / 32, 512 / 32, 4), blk(32, 8, 1);
    transpose_cvt<<<g, blk, 0, stream>>>(fmap, fmapT);
  }
  cvt(Wq, WqB, 512 * 512, qscale);
  cvt(Wk, WkB, 512 * 2048, 1.0f);
  cvt(Wv, WvB, 512 * 2048, 1.0f);
  cvt(Wout, WoutB, 512 * 512, 1.0f);
  im2col_s2k2<<<(4 * 1024 * 2048) / 256, 256, 0, stream>>>(fmap, imT);

  {  // Qt = fmapT * Wq^T
    dim3 g(4096 / 128, 512 / 256, 4);
    gemm_bf16<true, false><<<g, 256, 0, stream>>>(
        fmapT, WqB, Qt, 4096, 512, 512, 4096l * 512, 0, 4096l * 512);
  }
  {  // Kt = imT * Wk^T ; Vt = (imT * Wv^T)^T
    dim3 g(1024 / 128, 512 / 256, 4);
    gemm_bf16<true, false><<<g, 256, 0, stream>>>(
        imT, WkB, Ktb, 1024, 512, 2048, 1024l * 2048, 0, 1024l * 512);
    gemm_bf16<true, true><<<g, 256, 0, stream>>>(
        imT, WvB, Vtb, 1024, 512, 2048, 1024l * 2048, 0, 512l * 1024);
  }
  {  // attention
    dim3 g(64, 32, 1);
    flash_attn<<<g, 128, 0, stream>>>(Qt, Ktb, Vtb, nullkv, AO);
  }
  {  // out = Wout * AO^T -> f32 [b][512][4096]
    dim3 g(512 / 128, 4096 / 256, 4);
    gemm_bf16<false, false><<<g, 256, 0, stream>>>(
        WoutB, AO, out, 512, 4096, 512, 0, 4096l * 512, 512l * 4096);
  }
}
